// LowRank_82549271429294
// MI455X (gfx1250) — compile-verified, matrix operand order checked by reading, not run on hardware
//
#include <hip/hip_runtime.h>
#include <hip/hip_bf16.h>
#include <cstdint>

#define BB   128
#define MM   1024
#define EE   1024
#define HH   8
#define DD   128
#define MEMN 40
#define MIDN 64
#define LTOT (MM + MEMN)   /* 1064 */
#define ALPHA_C 1.3f
#define EPS_C   1e-5f

typedef __attribute__((ext_vector_type(16))) __bf16 v16bf;
typedef __attribute__((ext_vector_type(8)))  __bf16 bf16x8;
typedef __attribute__((ext_vector_type(4)))  __bf16 bf16x4;
typedef __attribute__((ext_vector_type(8)))  float  v8f;

// LDS tile geometry: 8 subtiles (16x32) x 32 lane-chunks; each chunk holds the
// 16 bf16 of one lane's WMMA fragment, padded to 24 elems (48B) so the 16-lane
// fragment b128 reads land on distinct bank groups.
#define CHUNK_E 24
#define TILE_E  (256 * CHUNK_E)   /* 6144 bf16 = 12 KB per tile */

__device__ __forceinline__ float celu_f(float x) {
    return x > 0.f ? x : ALPHA_C * (__expf(x * (1.0f / ALPHA_C)) - 1.0f);
}

// Stage a 128x32 f32 tile (rows at src + r*EE, cols k0..k0+31) into LDS in
// WMMA-fragment order, converting to bf16. Fragment element index for column
// k (0..31): half = (k>>3)&1 selects lane group, i = ((k>>4)<<3) + (k&7).
__device__ __forceinline__ void stage_tile(const float* __restrict__ src,
                                           __bf16* __restrict__ dst,
                                           int k0, int tid)
{
    const int sr   = tid >> 3;          // 0..31 (x4 passes)
    const int sc   = (tid & 7) * 4;     // 0,4,...,28 (float4 granularity)
    const int half = (sc >> 3) & 1;
    const int i0   = ((sc >> 4) << 3) + (sc & 7);
#pragma unroll
    for (int p = 0; p < 4; ++p) {
        const int r = sr + p * 32;
        const float4 x = *(const float4*)(src + (size_t)r * EE + k0 + sc);
        const int chunk = ((r >> 4) << 5) + (r & 15) + (half << 4);
        bf16x4 v;
        v[0] = (__bf16)x.x; v[1] = (__bf16)x.y;
        v[2] = (__bf16)x.z; v[3] = (__bf16)x.w;
        *(bf16x4*)&dst[chunk * CHUNK_E + i0] = v;   // one 8B DS store
    }
}

__device__ __forceinline__ v16bf load_frag(const __bf16* __restrict__ p) {
    const bf16x8 lo = *(const bf16x8*)(p);          // ds_load_b128
    const bf16x8 hi = *(const bf16x8*)(p + 8);      // ds_load_b128
    v16bf f;
#pragma unroll
    for (int i = 0; i < 8; ++i) { f[i] = lo[i]; f[i + 8] = hi[i]; }
    return f;
}

// -------------------------------------------------------------------------
// Fused projection: out = groupnorm(celu(X @ W^T + bias)) -> bf16.
// Grid: (E/128, nrows/128)  [col-block FAST -> X rows reused in L2].
// Block: 256 threads = 8 waves; wave w computes rows w*16..w*16+15 x 128 cols
// via 8 WMMA accumulators; N tile == one groupnorm group (D=128).
// -------------------------------------------------------------------------
__global__ __launch_bounds__(256)
void proj_gn_kernel(const float* __restrict__ X, const float* __restrict__ W,
                    const float* __restrict__ bias, const float* __restrict__ gamma,
                    const float* __restrict__ beta, __bf16* __restrict__ out,
                    int nrows)
{
    (void)nrows;
    __shared__ __bf16 As[2][TILE_E];
    __shared__ __bf16 Bs[2][TILE_E];

    const int tid  = threadIdx.x;
    const int lane = tid & 31;
    const int wave = tid >> 5;
    const int n0   = blockIdx.x * 128;   // fast dim: column block (head group)
    const int row0 = blockIdx.y * 128;

    const int half = lane >> 4;
    const int l16  = lane & 15;

    const v8f vzero = {0.f, 0.f, 0.f, 0.f, 0.f, 0.f, 0.f, 0.f};
    v8f acc[8];
#pragma unroll
    for (int t = 0; t < 8; ++t) acc[t] = vzero;

    const float* Xb = X + (size_t)row0 * EE;
    const float* Wb = W + (size_t)n0 * EE;

    // prologue: stage k0 = 0 into buffer 0
    stage_tile(Xb, As[0], 0, tid);
    stage_tile(Wb, Bs[0], 0, tid);

    int cur = 0;
    for (int k0 = 0; k0 < EE; k0 += 32) {
        __syncthreads();
        const int nxt = cur ^ 1;
        if (k0 + 32 < EE) {               // overlap next-tile staging with WMMA
            stage_tile(Xb, As[nxt], k0 + 32, tid);
            stage_tile(Wb, Bs[nxt], k0 + 32, tid);
        }
        const __bf16* ap = As[cur];
        const __bf16* bp = Bs[cur];

        const v16bf af = load_frag(&ap[(wave * 32 + lane) * CHUNK_E]);
        v16bf bfr[8];
#pragma unroll
        for (int nt = 0; nt < 8; ++nt)
            bfr[nt] = load_frag(&bp[(nt * 32 + lane) * CHUNK_E]);
#pragma unroll
        for (int nt = 0; nt < 8; ++nt)
            acc[nt] = __builtin_amdgcn_wmma_f32_16x16x32_bf16(
                false, af, false, bfr[nt], (short)0, acc[nt], false, false);
        cur = nxt;
    }

    // Epilogue: bias -> celu -> groupnorm over this 128-wide group
    float bcol[8], gcol[8], scol[8];
#pragma unroll
    for (int nt = 0; nt < 8; ++nt) {
        const int col = n0 + nt * 16 + l16;
        bcol[nt] = bias[col]; gcol[nt] = gamma[col]; scol[nt] = beta[col];
    }
#pragma unroll
    for (int r = 0; r < 8; ++r) {
        // C layout: lanes 0-15 hold M=r, lanes 16-31 hold M=r+8
        const int row = row0 + wave * 16 + r + half * 8;
        float v[8];
        float s = 0.f, s2 = 0.f;
#pragma unroll
        for (int nt = 0; nt < 8; ++nt) {
            float x = acc[nt][r] + bcol[nt];
            x = celu_f(x);
            v[nt] = x; s += x; s2 += x * x;
        }
#pragma unroll
        for (int off = 1; off < 16; off <<= 1) {
            s  += __shfl_xor(s,  off, 16);
            s2 += __shfl_xor(s2, off, 16);
        }
        const float mean = s * (1.0f / 128.0f);
        const float var  = s2 * (1.0f / 128.0f) - mean * mean;
        const float rstd = rsqrtf(var + EPS_C);
#pragma unroll
        for (int nt = 0; nt < 8; ++nt) {
            const float y = (v[nt] - mean) * rstd * gcol[nt] + scol[nt];
            out[(size_t)row * EE + (n0 + nt * 16 + l16)] = (__bf16)y;
        }
    }
}

// -------------------------------------------------------------------------
// Attention stage: one workgroup per (b,h). 256 threads = 8 waves.
// -------------------------------------------------------------------------
__global__ __launch_bounds__(256)
void attn_kernel(const __bf16* __restrict__ q_bf, const __bf16* __restrict__ v1_bf,
                 const __bf16* __restrict__ k_bf, const __bf16* __restrict__ v2_bf,
                 const float* __restrict__ mem,  const float* __restrict__ mask,
                 const float* __restrict__ Wb,   const float* __restrict__ bb,
                 const float* __restrict__ Wl,   const float* __restrict__ bl,
                 const float* __restrict__ Wl2,  const float* __restrict__ bl2,
                 float* __restrict__ out)
{
    __shared__ float WbT_s[128 * 65];   // [d][o], stride 65 -> conflict-free
    __shared__ float q_s[128];
    __shared__ float Wl_s[64];
    __shared__ float bb_s[64];
    __shared__ float a_s[8][128];       // per-wave q*k row
    __shared__ float logit_s[LTOT];
    __shared__ float pool_s[64];
    __shared__ float red_s[256];
    __shared__ float v2p_s[2][128];
    __shared__ float maxv_s, sum_s, msum_s;

    const int tid  = threadIdx.x;
    const int lane = tid & 31;
    const int wave = tid >> 5;
    const int bh   = blockIdx.x;
    const int b    = bh / HH;
    const int h    = bh % HH;
    const size_t qoff = (size_t)b * EE + (size_t)h * DD;

    for (int i = tid; i < MIDN * DD; i += 256) {
        const int o = i / DD, d = i % DD;
        WbT_s[d * 65 + o] = Wb[i];          // Wb is [o][d] row-major
    }
    if (tid < DD)   q_s[tid] = (float)q_bf[qoff + tid];
    if (tid < MIDN) { Wl_s[tid] = Wl[tid]; bb_s[tid] = bb[tid]; pool_s[tid] = 0.f; }
    if (tid == 0)   msum_s = 0.f;
    __syncthreads();

    const float sqrtD   = 11.313708498984761f;  // sqrt(D=128)
    const float sqrtMEM = 6.324555320336759f;   // sqrt(MEM=40)
    const float blv = bl[0];

    const int o1 = lane, o2 = lane + 32;
    float pAcc1 = 0.f, pAcc2 = 0.f, mAcc = 0.f;

    // 1064 = 8 waves * 133 iterations -> uniform barrier count
    for (int t = 0; t < LTOT / 8; ++t) {
        const int m = wave + t * 8;
#pragma unroll
        for (int j = 0; j < 4; ++j) {
            const int d = lane + j * 32;
            float kv;
            if (m < MM) kv = (float)k_bf[((size_t)b * MM + m) * EE + h * DD + d];
            else        kv = sqrtD * mem[(size_t)(m - MM) * EE + h * DD + d];
            a_s[wave][d] = q_s[d] * kv;
        }
        __syncthreads();

        float h1 = bb_s[o1], h2 = bb_s[o2];
#pragma unroll 4
        for (int d = 0; d < DD; ++d) {
            const float a = a_s[wave][d];
            h1 = __fmaf_rn(a, WbT_s[d * 65 + o1], h1);
            h2 = __fmaf_rn(a, WbT_s[d * 65 + o2], h2);
        }
        h1 = fmaxf(h1, 0.f); h2 = fmaxf(h2, 0.f);

        const float mv = mask[(size_t)b * MM + (m < MM ? m : m - MM)];
        pAcc1 += h1 * mv; pAcc2 += h2 * mv; mAcc += mv;

        float part = h1 * Wl_s[o1] + h2 * Wl_s[o2];
#pragma unroll
        for (int off = 16; off >= 1; off >>= 1) part += __shfl_xor(part, off, 32);
        if (lane == 0) logit_s[m] = (mv == 0.f) ? -1e9f : (part + blv);
    }

    atomicAdd(&pool_s[o1], pAcc1);
    atomicAdd(&pool_s[o2], pAcc2);
    if (lane == 0) atomicAdd(&msum_s, mAcc);
    __syncthreads();

    // softmax over logit_s[0..LTOT)
    float lm = -1e30f;
    for (int m = tid; m < LTOT; m += 256) lm = fmaxf(lm, logit_s[m]);
    red_s[tid] = lm; __syncthreads();
    if (tid == 0) {
        float mx = -1e30f;
        for (int i = 0; i < 256; ++i) mx = fmaxf(mx, red_s[i]);
        maxv_s = mx;
    }
    __syncthreads();
    const float mx = maxv_s;
    float ls = 0.f;
    for (int m = tid; m < LTOT; m += 256) {
        const float e = __expf(logit_s[m] - mx);
        logit_s[m] = e; ls += e;
    }
    red_s[tid] = ls; __syncthreads();
    if (tid == 0) {
        float ssum = 0.f;
        for (int i = 0; i < 256; ++i) ssum += red_s[i];
        sum_s = ssum;
    }
    __syncthreads();
    const float inv_sum = 1.0f / sum_s;

    // v2p[d] = sum_m p_m * v2_full[m][d]; two threads per d (interleaved m)
    {
        const int d = tid & 127, hf = tid >> 7;
        float acc = 0.f;
        for (int m = hf; m < LTOT; m += 2) {
            float val;
            if (m < MM) val = (float)v2_bf[((size_t)b * MM + m) * EE + h * DD + d];
            else        val = sqrtMEM * mem[(size_t)(m - MM) * EE + h * DD + d];
            acc += logit_s[m] * val;
        }
        v2p_s[hf][d] = acc;
    }
    __syncthreads();

    if (tid < DD) {
        const int d = tid;
        const float imsum = 1.0f / msum_s;
        float z = bl2[d];
#pragma unroll 4
        for (int o = 0; o < MIDN; ++o) z += (pool_s[o] * imsum) * Wl2[d * MIDN + o];
        const float ac  = 1.0f / (1.0f + __expf(-z));
        const float v2p = (v2p_s[0][d] + v2p_s[1][d]) * inv_sum;
        const float v1v = (float)v1_bf[qoff + d];
        out[qoff + d] = v1v * v2p * ac;
    }
}

// -------------------------------------------------------------------------
extern "C" void kernel_launch(void* const* d_in, const int* in_sizes, int n_in,
                              void* d_out, int out_size, void* d_ws, size_t ws_size,
                              hipStream_t stream)
{
    (void)in_sizes; (void)n_in; (void)out_size; (void)ws_size;
    const float* query  = (const float*)d_in[0];
    const float* key    = (const float*)d_in[1];
    const float* mask   = (const float*)d_in[2];
    const float* value1 = (const float*)d_in[3];
    const float* value2 = (const float*)d_in[4];
    const float* Wq  = (const float*)d_in[5];  const float* bq  = (const float*)d_in[6];
    const float* gq  = (const float*)d_in[7];  const float* sq  = (const float*)d_in[8];
    const float* Wk  = (const float*)d_in[9];  const float* bk  = (const float*)d_in[10];
    const float* gk  = (const float*)d_in[11]; const float* sk  = (const float*)d_in[12];
    const float* Wv1 = (const float*)d_in[13]; const float* bv1 = (const float*)d_in[14];
    const float* gv1 = (const float*)d_in[15]; const float* sv1 = (const float*)d_in[16];
    const float* Wv2 = (const float*)d_in[17]; const float* bv2 = (const float*)d_in[18];
    const float* gv2 = (const float*)d_in[19]; const float* sv2 = (const float*)d_in[20];
    const float* mem = (const float*)d_in[21];
    const float* Wb  = (const float*)d_in[22]; const float* bb  = (const float*)d_in[23];
    const float* Wl  = (const float*)d_in[24]; const float* bl  = (const float*)d_in[25];
    const float* Wl2 = (const float*)d_in[26]; const float* bl2 = (const float*)d_in[27];

    // Workspace layout (bf16 intermediates): ~512.5 MB total
    char* ws = (char*)d_ws;
    __bf16* q_bf  = (__bf16*)ws;  ws += (size_t)BB * EE * 2;
    __bf16* v1_bf = (__bf16*)ws;  ws += (size_t)BB * EE * 2;
    __bf16* k_bf  = (__bf16*)ws;  ws += (size_t)BB * MM * EE * 2;
    __bf16* v2_bf = (__bf16*)ws;

    const dim3 blk(256);
    // grid.x = column blocks (8) -> consecutive workgroups share X rows in L2
    proj_gn_kernel<<<dim3(EE / 128, BB / 128), blk, 0, stream>>>(
        query,  Wq,  bq,  gq,  sq,  q_bf,  BB);
    proj_gn_kernel<<<dim3(EE / 128, BB / 128), blk, 0, stream>>>(
        value1, Wv1, bv1, gv1, sv1, v1_bf, BB);
    proj_gn_kernel<<<dim3(EE / 128, (BB * MM) / 128), blk, 0, stream>>>(
        key,    Wk,  bk,  gk,  sk,  k_bf,  BB * MM);
    proj_gn_kernel<<<dim3(EE / 128, (BB * MM) / 128), blk, 0, stream>>>(
        value2, Wv2, bv2, gv2, sv2, v2_bf, BB * MM);

    attn_kernel<<<dim3(BB * HH), blk, 0, stream>>>(
        q_bf, v1_bf, k_bf, v2_bf, mem, mask, Wb, bb, Wl, bl, Wl2, bl2,
        (float*)d_out);
}
